// EtudeRNN_59820304498865
// MI455X (gfx1250) — matrix operand order repaired and checked by reading, not compiled, not run on hardware
//
#include <hip/hip_runtime.h>
#include <hip/hip_bf16.h>

typedef __attribute__((ext_vector_type(16))) __bf16 v16bf;
typedef __attribute__((ext_vector_type(8)))  __bf16 v8bf;
typedef __attribute__((ext_vector_type(8)))  float  v8f;

#define DEVFN static __device__ __forceinline__

// Problem constants (reference: L=3, B=64, H=1024, D=128, P=64, T=512)
constexpr int kL = 3, kB = 64, kH = 1024, kD = 128, kP = 64, kT = 512;
constexpr int k4H = 4 * kH;          // 4096 gate columns
constexpr int kBH = kB * kH;         // 65536
constexpr int kBD = kB * kD;         // 8192

DEVFN float sigmoidf_(float x) { return 1.0f / (1.0f + __expf(-x)); }

DEVFN v8f wmma_bf16(v16bf a, v16bf b, v8f c) {
  // D = A(16x32 bf16) * B(32x16 bf16) + C(16x16 f32)
  return __builtin_amdgcn_wmma_f32_16x16x32_bf16(false, a, false, b, (short)0, c,
                                                 false, false);
}

// A-fragment (16x32 bf16, row-major source): this lane covers row m = lane%16,
// K groups [kb..kb+7] and [kb+16..kb+23], kb = (lane&16 ? 8 : 0). Two aligned
// 16-byte loads merged into one v16bf.
DEVFN v16bf load_a_frag(const __bf16* rowPtr) {
  v8bf lo = *(const v8bf*)(rowPtr);
  v8bf hi = *(const v8bf*)(rowPtr + 16);
  return __builtin_shufflevector(lo, hi, 0, 1, 2, 3, 4, 5, 6, 7,
                                          8, 9, 10, 11, 12, 13, 14, 15);
}

// ---------------------------------------------------------------------------
// Fused LSTM layer step: gates = Xin @ WtIh + Hin @ WtHh (+bias), then cell.
//   Xin   : B x Kin  bf16 (row-major)
//   WtIh  : 4H x Kin bf16 (row-major, row = gate column)
//   WtHh  : 4H x H   bf16 (row-major, row = gate column)
//   HinBf : B x H bf16 (read slot), HoutBf : B x H bf16 (write slot)
//   Cst   : B x H f32 (in-place; only (m,j) owned by this block touched)
// Grid: H/32 blocks, 256 threads (8 waves). Wave w -> gate g=w>>1, 16-col tile.
// ---------------------------------------------------------------------------
__global__ __launch_bounds__(256)
void k_lstm_layer(const __bf16* __restrict__ Xin, int Kin,
                  const __bf16* __restrict__ WtIh,
                  const __bf16* __restrict__ WtHh,
                  const __bf16* __restrict__ HinBf,
                  __bf16* __restrict__ HoutBf,
                  float* __restrict__ Cst,
                  const float* __restrict__ bias) {
  __shared__ float gbuf[4][kB][32];

  const int tid   = threadIdx.x;
  const int wave  = tid >> 5, lane = tid & 31;
  const int g     = wave >> 1, jt = wave & 1;
  const int jcol0 = blockIdx.x * 32;
  const int ncol  = g * kH + jcol0 + jt * 16;  // gate-space column of this wave
  const int mlo   = lane & 15;
  const int kboff = (lane & 16) ? 8 : 0;

  v8f acc[4] = {};

  // Input GEMM: K = Kin
  for (int k0 = 0; k0 < Kin; k0 += 32) {
    v16bf bf = load_a_frag(WtIh + (size_t)(ncol + mlo) * Kin + k0 + kboff);
#pragma unroll
    for (int mt = 0; mt < 4; ++mt) {
      v16bf af = load_a_frag(Xin + (size_t)(mt * 16 + mlo) * Kin + k0 + kboff);
      acc[mt] = wmma_bf16(af, bf, acc[mt]);
    }
  }
  // Recurrent GEMM: K = H
  for (int k0 = 0; k0 < kH; k0 += 32) {
    v16bf bf = load_a_frag(WtHh + (size_t)(ncol + mlo) * kH + k0 + kboff);
#pragma unroll
    for (int mt = 0; mt < 4; ++mt) {
      v16bf af = load_a_frag(HinBf + (size_t)(mt * 16 + mlo) * kH + k0 + kboff);
      acc[mt] = wmma_bf16(af, bf, acc[mt]);
    }
  }

  // Spill gate tiles to LDS so all 4 gates of a (b,j) meet in one thread.
  const int jl16 = jt * 16 + mlo;
  const int mhi  = (lane & 16) ? 8 : 0;
#pragma unroll
  for (int mt = 0; mt < 4; ++mt)
#pragma unroll
    for (int v = 0; v < 8; ++v)
      gbuf[g][mt * 16 + v + mhi][jl16] = acc[mt][v];

  __syncthreads();

  // Fused LSTM pointwise (f32 state, bf16 hidden out).
  for (int idx = tid; idx < kB * 32; idx += 256) {
    const int m = idx >> 5, jl = idx & 31;
    const int j = jcol0 + jl;
    const float ig = sigmoidf_(gbuf[0][m][jl] + bias[j]);
    const float fg = sigmoidf_(gbuf[1][m][jl] + bias[kH + j]);
    const float gg = tanhf(gbuf[2][m][jl] + bias[2 * kH + j]);
    const float og = sigmoidf_(gbuf[3][m][jl] + bias[3 * kH + j]);
    const size_t o = (size_t)m * kH + j;
    const float cn = fg * Cst[o] + ig * gg;
    Cst[o] = cn;
    HoutBf[o] = (__bf16)(og * tanhf(cn));
  }
}

// ---------------------------------------------------------------------------
// FC + sigmoid + feedback: gen = sigmoid(H2 @ WtFc + bfc); writes fp32 output
// row and bf16 next input. One block, 8 waves (wave = 16-col N tile).
// ---------------------------------------------------------------------------
__global__ __launch_bounds__(256)
void k_fc_gen(const __bf16* __restrict__ H2,
              const __bf16* __restrict__ WtFc,  // D x H bf16 (row = output col)
              const float* __restrict__ bfc,
              float* __restrict__ outRow,       // B x D fp32
              __bf16* __restrict__ XinNext) {   // B x D bf16
  const int tid = threadIdx.x;
  const int wave = tid >> 5, lane = tid & 31;
  const int n0 = wave * 16;
  const int mlo = lane & 15;
  const int kboff = (lane & 16) ? 8 : 0;

  v8f acc[4] = {};
  for (int k0 = 0; k0 < kH; k0 += 32) {
    v16bf bf = load_a_frag(WtFc + (size_t)(n0 + mlo) * kH + k0 + kboff);
#pragma unroll
    for (int mt = 0; mt < 4; ++mt) {
      v16bf af = load_a_frag(H2 + (size_t)(mt * 16 + mlo) * kH + k0 + kboff);
      acc[mt] = wmma_bf16(af, bf, acc[mt]);
    }
  }
  const int n = n0 + mlo;
  const float bb = bfc[n];
  const int mhi = (lane & 16) ? 8 : 0;
#pragma unroll
  for (int mt = 0; mt < 4; ++mt)
#pragma unroll
    for (int v = 0; v < 8; ++v) {
      const int m = mt * 16 + v + mhi;
      const float s = sigmoidf_(acc[mt][v] + bb);
      outRow[m * kD + n] = s;
      XinNext[m * kD + n] = (__bf16)s;
    }
}

// ---------------------------------------------------------------------------
// One-time setup kernels
// ---------------------------------------------------------------------------
// dst (N x K bf16) = convert of src (N x K f32)
__global__ void k_transpose_convert(const float* __restrict__ src,
                                    __bf16* __restrict__ dst, int N, int K) {
  const int i = blockIdx.x * blockDim.x + threadIdx.x;
  if (i >= N * K) return;
  const int k = i / N, n = i - k * N;
  dst[(size_t)n * K + k] = (__bf16)src[(size_t)n * K + k];
}

__global__ void k_init_bias(const float* bi0, const float* bh0,
                            const float* bi1, const float* bh1,
                            const float* bi2, const float* bh2,
                            const float* bfs, float* b0, float* b1, float* b2,
                            float* bfc) {
  const int i = blockIdx.x * blockDim.x + threadIdx.x;
  if (i < k4H) {
    b0[i] = bi0[i] + bh0[i];
    b1[i] = bi1[i] + bh1[i];
    b2[i] = bi2[i] + bh2[i];
  }
  if (i < kD) bfc[i] = bfs[i];
}

__global__ void k_init_states(const float* __restrict__ h0,
                              const float* __restrict__ c0,
                              __bf16* __restrict__ hbf,
                              float* __restrict__ cst) {
  const int i = blockIdx.x * blockDim.x + threadIdx.x;
  if (i >= kL * kBH) return;
  cst[i] = c0[i];
  const int l = i / kBH, r = i - l * kBH;
  hbf[(size_t)(l * 2) * kBH + r] = (__bf16)h0[i];  // slot 0
}

__global__ void k_init_prompt(const float* __restrict__ x,
                              __bf16* __restrict__ xbf,
                              __bf16* __restrict__ xin,
                              float* __restrict__ outp) {
  const int i = blockIdx.x * blockDim.x + threadIdx.x;
  if (i >= kP * kBD) return;
  const float v = x[i];
  xbf[i] = (__bf16)v;
  outp[i] = v;                       // out[0..P-1] = prompt
  if (i < kBD) xin[i] = (__bf16)v;   // first input = x[0]
}

__global__ void k_copy_xin(const __bf16* __restrict__ src,
                           __bf16* __restrict__ xin) {
  const int i = blockIdx.x * blockDim.x + threadIdx.x;
  if (i < kBD) xin[i] = src[i];
}

// ---------------------------------------------------------------------------
extern "C" void kernel_launch(void* const* d_in, const int* in_sizes, int n_in,
                              void* d_out, int out_size, void* d_ws,
                              size_t ws_size, hipStream_t stream) {
  const float* x    = (const float*)d_in[0];
  const float* h0   = (const float*)d_in[1];
  const float* c0   = (const float*)d_in[2];
  const float* Wih0 = (const float*)d_in[3];
  const float* Whh0 = (const float*)d_in[4];
  const float* bih0 = (const float*)d_in[5];
  const float* bhh0 = (const float*)d_in[6];
  const float* Wih1 = (const float*)d_in[7];
  const float* Whh1 = (const float*)d_in[8];
  const float* bih1 = (const float*)d_in[9];
  const float* bhh1 = (const float*)d_in[10];
  const float* Wih2 = (const float*)d_in[11];
  const float* Whh2 = (const float*)d_in[12];
  const float* bih2 = (const float*)d_in[13];
  const float* bhh2 = (const float*)d_in[14];
  const float* Wfc  = (const float*)d_in[15];
  const float* bfcS = (const float*)d_in[16];
  // d_in[17] = timesteps (== 512, baked into kT)
  float* out = (float*)d_out;

  // Workspace carve (~44 MB): bf16 weights fit in the 192 MB L2.
  char* cur = (char*)d_ws;
  auto carve = [&](size_t bytes) {
    char* p = cur;
    cur += (bytes + 255) & ~(size_t)255;
    return p;
  };
  __bf16* WtIh0 = (__bf16*)carve((size_t)kD * k4H * 2);
  __bf16* WtHh0 = (__bf16*)carve((size_t)kH * k4H * 2);
  __bf16* WtIh1 = (__bf16*)carve((size_t)kH * k4H * 2);
  __bf16* WtHh1 = (__bf16*)carve((size_t)kH * k4H * 2);
  __bf16* WtIh2 = (__bf16*)carve((size_t)kH * k4H * 2);
  __bf16* WtHh2 = (__bf16*)carve((size_t)kH * k4H * 2);
  __bf16* WtFc  = (__bf16*)carve((size_t)kH * kD * 2);
  float*  b0    = (float*)carve((size_t)k4H * 4);
  float*  b1    = (float*)carve((size_t)k4H * 4);
  float*  b2    = (float*)carve((size_t)k4H * 4);
  float*  bfc   = (float*)carve((size_t)kD * 4);
  __bf16* hbf   = (__bf16*)carve((size_t)kL * 2 * kBH * 2);  // [layer][slot]
  float*  cst   = (float*)carve((size_t)kL * kBH * 4);
  __bf16* xbf   = (__bf16*)carve((size_t)kP * kBD * 2);
  __bf16* xin   = (__bf16*)carve((size_t)kBD * 2);

  auto gridFor = [](size_t n) { return (unsigned)((n + 255) / 256); };

  // One-time: convert weights to bf16 N x K (no transpose), biases, states, prompt.
  k_transpose_convert<<<gridFor((size_t)k4H * kD), 256, 0, stream>>>(Wih0, WtIh0, k4H, kD);
  k_transpose_convert<<<gridFor((size_t)k4H * kH), 256, 0, stream>>>(Whh0, WtHh0, k4H, kH);
  k_transpose_convert<<<gridFor((size_t)k4H * kH), 256, 0, stream>>>(Wih1, WtIh1, k4H, kH);
  k_transpose_convert<<<gridFor((size_t)k4H * kH), 256, 0, stream>>>(Whh1, WtHh1, k4H, kH);
  k_transpose_convert<<<gridFor((size_t)k4H * kH), 256, 0, stream>>>(Wih2, WtIh2, k4H, kH);
  k_transpose_convert<<<gridFor((size_t)k4H * kH), 256, 0, stream>>>(Whh2, WtHh2, k4H, kH);
  k_transpose_convert<<<gridFor((size_t)kD * kH), 256, 0, stream>>>(Wfc, WtFc, kD, kH);
  k_init_bias<<<gridFor(k4H), 256, 0, stream>>>(bih0, bhh0, bih1, bhh1, bih2,
                                                bhh2, bfcS, b0, b1, b2, bfc);
  k_init_states<<<gridFor((size_t)kL * kBH), 256, 0, stream>>>(h0, c0, hbf, cst);
  k_init_prompt<<<gridFor((size_t)kP * kBD), 256, 0, stream>>>(x, xbf, xin, out);

  // Sequential rollout. Hidden state double-buffered (slot = t parity) so a
  // block never reads columns another block is rewriting within a step.
  for (int t = 0; t < kT - 1; ++t) {
    const int si = t & 1, so = si ^ 1;
    __bf16* h0i = hbf + (size_t)(0 * 2 + si) * kBH;
    __bf16* h0o = hbf + (size_t)(0 * 2 + so) * kBH;
    __bf16* h1i = hbf + (size_t)(1 * 2 + si) * kBH;
    __bf16* h1o = hbf + (size_t)(1 * 2 + so) * kBH;
    __bf16* h2i = hbf + (size_t)(2 * 2 + si) * kBH;
    __bf16* h2o = hbf + (size_t)(2 * 2 + so) * kBH;

    k_lstm_layer<<<kH / 32, 256, 0, stream>>>(xin, kD, WtIh0, WtHh0, h0i, h0o,
                                              cst + 0 * (size_t)kBH, b0);
    k_lstm_layer<<<kH / 32, 256, 0, stream>>>(h0o, kH, WtIh1, WtHh1, h1i, h1o,
                                              cst + 1 * (size_t)kBH, b1);
    k_lstm_layer<<<kH / 32, 256, 0, stream>>>(h1o, kH, WtIh2, WtHh2, h2i, h2o,
                                              cst + 2 * (size_t)kBH, b2);
    const int tp1 = t + 1;
    if (tp1 < kP) {
      // Teacher forcing: next input is the prompt; output row already written.
      k_copy_xin<<<gridFor(kBD), 256, 0, stream>>>(xbf + (size_t)tp1 * kBD, xin);
    } else {
      k_fc_gen<<<1, 256, 0, stream>>>(h2o, WtFc, bfc,
                                      out + (size_t)tp1 * kBD, xin);
    }
  }
  (void)in_sizes; (void)n_in; (void)out_size; (void)ws_size;
}
